// AddTreeLSTM_26087631356092
// MI455X (gfx1250) — matrix operand order repaired and checked by reading, not compiled, not run on hardware
//
#include <hip/hip_runtime.h>

#define N_NODES   4096
#define IN_SIZE   1024
#define EDGE_SIZE 128
#define HID       1024
#define THREE_H   3072
#define D_IN      1152
#define OUT_COLS  4096   // THREE_H + HID, fused output width
#define SCAN_BLOCKS  96
#define SCAN_THREADS 256
#define MAX_KIDS  16

typedef __attribute__((ext_vector_type(16))) __bf16 v16bf;
typedef __attribute__((ext_vector_type(8)))  float  v8f;

__device__ __forceinline__ unsigned short f2bf_bits(float f) {
  union { float f; unsigned u; } in; in.f = f;
  unsigned r = in.u + 0x7FFFu + ((in.u >> 16) & 1u);   // round-nearest-even
  return (unsigned short)(r >> 16);
}
// unpack packed pair of bf16 from a 32-bit word
__device__ __forceinline__ float bflo(unsigned u) {
  union { unsigned u; float f; } v; v.u = u << 16; return v.f;
}
__device__ __forceinline__ float bfhi(unsigned u) {
  union { unsigned u; float f; } v; v.u = u & 0xFFFF0000u; return v.f;
}
__device__ __forceinline__ float sigm(float x) { return 1.0f / (1.0f + __expf(-x)); }

// ---------------------------------------------------------------------------
// Prep kernels: one-time f32 -> bf16 packing (removes all conversion VALU
// from the WMMA inner loop and halves its load bytes).
// ---------------------------------------------------------------------------
__global__ void pack_bf16(const float* __restrict__ src,
                          unsigned short* __restrict__ dst, int n) {
  int i = blockIdx.x * blockDim.x + threadIdx.x;
  if (i < n) dst[i] = f2bf_bits(src[i]);
}

// seq16[r, 0:1024] = inputs[r], seq16[r, 1024:1152] = edge[r]; also reset bar.
__global__ void build_seq16(const float* __restrict__ inputs,
                            const float* __restrict__ edge,
                            unsigned short* __restrict__ seq16,
                            unsigned* __restrict__ bar) {
  int i = blockIdx.x * blockDim.x + threadIdx.x;
  if (i == 0) { bar[0] = 0u; bar[1] = 0u; }
  if (i >= N_NODES * D_IN) return;
  int r = i / D_IN;
  int c = i - r * D_IN;
  float v = (c < IN_SIZE) ? inputs[(size_t)r * IN_SIZE + c]
                          : edge[(size_t)r * EDGE_SIZE + (c - IN_SIZE)];
  seq16[i] = f2bf_bits(v);
}

// ---------------------------------------------------------------------------
// Fused input GEMM via WMMA bf16 on pre-packed operands:
//   out[N, 4096] = seq16[N,1152] @ Wcat16[4096,1152]^T + [b_ioux ; b_fx]
// Block = 256 threads = 8 waves; wave computes 16 rows x 64 cols (4 tiles).
// ---------------------------------------------------------------------------
__global__ __launch_bounds__(256) void fused_input_gemm(
    const unsigned short* __restrict__ seq16,    // bf16 [N, 1152]
    const unsigned short* __restrict__ Wcat16,   // bf16 [4096, 1152]
    const float* __restrict__ b_ioux, const float* __restrict__ b_fx,
    float* __restrict__ out) {
  const int wave    = threadIdx.x >> 5;
  const int lane    = threadIdx.x & 31;
  const int l15     = lane & 15;
  const bool laneLow = lane < 16;
  const int rowBase = blockIdx.y * 128 + wave * 16;
  const int colBase = blockIdx.x * 64;

  union Frag { v16bf v; uint4 q[2]; };

  // Each lane owns one weight row (= one output column) per tile.
  const unsigned short* bRowPtr[4];
  float biasv[4];
#pragma unroll
  for (int j = 0; j < 4; ++j) {
    int nrow = colBase + j * 16 + l15;
    bRowPtr[j] = Wcat16 + (size_t)nrow * D_IN;
    biasv[j] = (nrow < THREE_H) ? b_ioux[nrow] : b_fx[nrow - THREE_H];
  }
  const unsigned short* aRowPtr = seq16 + (size_t)(rowBase + l15) * D_IN;

  v8f acc[4];
#pragma unroll
  for (int j = 0; j < 4; ++j)
#pragma unroll
    for (int r = 0; r < 8; ++r) acc[j][r] = 0.0f;

  for (int k0 = 0; k0 < D_IN; k0 += 32) {
    // A fragment, 16x32 bf16 ISA layout:
    //   lanes 0-15 : K = k0+0..7 and k0+16..23
    //   lanes 16-31: K = k0+8..15 and k0+24..31
    const int kA = k0 + (laneLow ? 0 : 8);
    Frag a;
    a.q[0] = *(const uint4*)(aRowPtr + kA);
    a.q[1] = *(const uint4*)(aRowPtr + kA + 16);

    // B fragment, 32x16: lane n<16 holds col n K=k0+0..7 and k0+16..23,
    // lane n+16 holds col n K=k0+8..15 and k0+24..31.
    const int kB = k0 + (laneLow ? 0 : 8);
#pragma unroll
    for (int j = 0; j < 4; ++j) {
      Frag b;
      b.q[0] = *(const uint4*)(bRowPtr[j] + kB);
      b.q[1] = *(const uint4*)(bRowPtr[j] + kB + 16);
      acc[j] = __builtin_amdgcn_wmma_f32_16x16x32_bf16(
          false, a.v, false, b.v, (short)0, acc[j], false, false);
    }
  }

  // C/D layout: VGPR r -> (lanes 0-15: M=r, N=lane) (lanes 16-31: M=r+8)
  const int mBase = rowBase + (laneLow ? 0 : 8);
#pragma unroll
  for (int j = 0; j < 4; ++j) {
    const int n = colBase + j * 16 + l15;
#pragma unroll
    for (int r = 0; r < 8; ++r)
      out[(size_t)(mBase + r) * OUT_COLS + n] = acc[j][r] + biasv[j];
  }
}

// ---------------------------------------------------------------------------
// Device-wide barrier for the persistent scan kernel
// ---------------------------------------------------------------------------
__device__ __forceinline__ void grid_barrier(unsigned* cnt, unsigned* gen, unsigned nb) {
  __threadfence();
  __syncthreads();
  if (threadIdx.x == 0) {
    unsigned g = __hip_atomic_load(gen, __ATOMIC_ACQUIRE, __HIP_MEMORY_SCOPE_AGENT);
    unsigned arrived = __hip_atomic_fetch_add(cnt, 1u, __ATOMIC_ACQ_REL, __HIP_MEMORY_SCOPE_AGENT);
    if (arrived == nb - 1) {
      __hip_atomic_store(cnt, 0u, __ATOMIC_RELAXED, __HIP_MEMORY_SCOPE_AGENT);
      __hip_atomic_fetch_add(gen, 1u, __ATOMIC_ACQ_REL, __HIP_MEMORY_SCOPE_AGENT);
    } else {
      while (__hip_atomic_load(gen, __ATOMIC_ACQUIRE, __HIP_MEMORY_SCOPE_AGENT) == g)
        __builtin_amdgcn_s_sleep(2);
    }
  }
  __syncthreads();
}

// ---------------------------------------------------------------------------
// Persistent sequential TreeLSTM scan. Per step:
//   (a) every block builds hsum in LDS,
//   (b) all waves in the grid do wave-per-row bf16 GEMV
//       (3072 iou rows + nk*1024 forget rows),
//   (c) block 0 applies gates, writes C[t], H[t]. Two grid barriers per step.
// ---------------------------------------------------------------------------
__global__ __launch_bounds__(SCAN_THREADS) void tree_scan(
    const float* __restrict__ big, const int* __restrict__ children,
    const unsigned char* __restrict__ mask, int Mch,
    const unsigned short* __restrict__ Wh16, const unsigned short* __restrict__ Wf16,
    const float* __restrict__ b_iouh, const float* __restrict__ b_fh,
    float* __restrict__ C, float* __restrict__ H,
    float* __restrict__ iou_h, float* __restrict__ f_pre,
    unsigned* __restrict__ bar) {
  __shared__ float s_hsum[HID];
  __shared__ int   s_kids[MAX_KIDS];
  __shared__ int   s_nk;
  const int lane   = threadIdx.x & 31;
  const int gwave  = (blockIdx.x * SCAN_THREADS + threadIdx.x) >> 5;
  const int nwaves = (SCAN_BLOCKS * SCAN_THREADS) >> 5;

  for (int t = 0; t < N_NODES; ++t) {
    // ---- phase 0: gather valid children, build hsum (masked sum of child h)
    if (threadIdx.x == 0) {
      int nk = 0;
      for (int m = 0; m < Mch && nk < MAX_KIDS; ++m)
        if (mask[(size_t)t * Mch + m]) s_kids[nk++] = children[(size_t)t * Mch + m];
      s_nk = nk;
    }
    __syncthreads();
    const int nk = s_nk;
    for (int j = threadIdx.x; j < HID; j += SCAN_THREADS) {
      float s = 0.0f;
      for (int m = 0; m < nk; ++m) s += H[(size_t)s_kids[m] * HID + j];
      s_hsum[j] = s;
    }
    __syncthreads();

    // ---- phase 1: wave-per-row GEMV over bf16 weights (L2-resident)
    const int totalRows = THREE_H + nk * HID;
    for (int r = gwave; r < totalRows; r += nwaves) {
      // prefetch next assigned row (32 lanes x 64B covers the 2KB row)
      int rn = r + nwaves;
      if (rn < totalRows) {
        const unsigned short* pw = (rn < THREE_H)
            ? (Wh16 + (size_t)rn * HID)
            : (Wf16 + (size_t)((rn - THREE_H) & (HID - 1)) * HID);
        __builtin_prefetch((const void*)(pw + lane * 32), 0, 0);
      }
      float accv = 0.0f;
      if (r < THREE_H) {
        const unsigned short* wr = Wh16 + (size_t)r * HID;
        for (int k = lane * 4; k < HID; k += 128) {
          uint2 w = *(const uint2*)(wr + k);
          float4 hv = *(const float4*)(s_hsum + k);
          accv += bflo(w.x) * hv.x + bfhi(w.x) * hv.y
                + bflo(w.y) * hv.z + bfhi(w.y) * hv.w;
        }
        for (int off = 16; off > 0; off >>= 1) accv += __shfl_down(accv, off, 32);
        if (lane == 0) iou_h[r] = accv + b_iouh[r];
      } else {
        int rr = r - THREE_H;
        int m = rr >> 10;
        int row = rr & (HID - 1);
        const unsigned short* wr = Wf16 + (size_t)row * HID;
        const float* hk = H + (size_t)s_kids[m] * HID;
        for (int k = lane * 4; k < HID; k += 128) {
          uint2 w = *(const uint2*)(wr + k);
          float4 hv = *(const float4*)(hk + k);
          accv += bflo(w.x) * hv.x + bfhi(w.x) * hv.y
                + bflo(w.y) * hv.z + bfhi(w.y) * hv.w;
        }
        for (int off = 16; off > 0; off >>= 1) accv += __shfl_down(accv, off, 32);
        if (lane == 0) f_pre[m * HID + row] = accv + b_fh[row];
      }
    }
    grid_barrier(bar, bar + 1, SCAN_BLOCKS);

    // ---- phase 2: gates + state update (block 0)
    if (blockIdx.x == 0) {
      const float* bigrow = big + (size_t)t * OUT_COLS;
      for (int j = threadIdx.x; j < HID; j += SCAN_THREADS) {
        float iv = sigm(bigrow[j]            + iou_h[j]);
        float ov = sigm(bigrow[HID + j]      + iou_h[HID + j]);
        float uv = tanhf(bigrow[2 * HID + j] + iou_h[2 * HID + j]);
        float fx = bigrow[THREE_H + j];
        float c = iv * uv;
        for (int m = 0; m < nk; ++m) {
          float fv = sigm(f_pre[m * HID + j] + fx);
          c += fv * C[(size_t)s_kids[m] * HID + j];
        }
        C[(size_t)t * HID + j] = c;
        H[(size_t)t * HID + j] = ov * tanhf(c);
      }
    }
    grid_barrier(bar, bar + 1, SCAN_BLOCKS);
  }
}

// ---------------------------------------------------------------------------
// Emit root (c, h) -> d_out (concatenated, 2048 f32)
// ---------------------------------------------------------------------------
__global__ void write_out(const float* __restrict__ C, const float* __restrict__ H,
                          float* __restrict__ out) {
  int j = blockIdx.x * blockDim.x + threadIdx.x;
  if (j < HID)            out[j] = C[(size_t)(N_NODES - 1) * HID + j];
  else if (j < 2 * HID)   out[j] = H[(size_t)(N_NODES - 1) * HID + (j - HID)];
}

// ---------------------------------------------------------------------------
extern "C" void kernel_launch(void* const* d_in, const int* in_sizes, int n_in,
                              void* d_out, int out_size, void* d_ws, size_t ws_size,
                              hipStream_t stream) {
  const float* inputs       = (const float*)d_in[0];
  const float* edge_inputs  = (const float*)d_in[1];
  const int* children       = (const int*)d_in[2];
  const unsigned char* mask = (const unsigned char*)d_in[3];
  const float* W_ioux       = (const float*)d_in[4];
  const float* b_ioux       = (const float*)d_in[5];
  const float* W_iouh       = (const float*)d_in[6];
  const float* b_iouh       = (const float*)d_in[7];
  const float* W_fx         = (const float*)d_in[8];
  const float* b_fx         = (const float*)d_in[9];
  const float* W_fh         = (const float*)d_in[10];
  const float* b_fh         = (const float*)d_in[11];

  const int Mch = in_sizes[2] / N_NODES;

  // Workspace carve (256B aligned).
  char* p = (char*)d_ws;
  auto carve = [&](size_t bytes) -> void* {
    void* r = (void*)p;
    p += (bytes + 255) & ~(size_t)255;
    return r;
  };
  float*          big    = (float*)carve((size_t)N_NODES * OUT_COLS * sizeof(float));      // 64 MB
  unsigned short* seq16  = (unsigned short*)carve((size_t)N_NODES * D_IN * 2);             // 9.4 MB
  unsigned short* Wcat16 = (unsigned short*)carve((size_t)OUT_COLS * D_IN * 2);            // 9.4 MB
  unsigned short* Wh16   = (unsigned short*)carve((size_t)THREE_H * HID * 2);              // 6 MB
  unsigned short* Wf16   = (unsigned short*)carve((size_t)HID * HID * 2);                  // 2 MB
  float*          C      = (float*)carve((size_t)N_NODES * HID * sizeof(float));           // 16 MB
  float*          H      = (float*)carve((size_t)N_NODES * HID * sizeof(float));           // 16 MB
  float*          iou_h  = (float*)carve((size_t)THREE_H * sizeof(float));
  float*          f_pre  = (float*)carve((size_t)MAX_KIDS * HID * sizeof(float));
  unsigned*       bar    = (unsigned*)carve(256);

  // 1) one-time bf16 packing + barrier reset (deterministic per call)
  {
    int n;
    n = THREE_H * D_IN;               // W_ioux -> Wcat16[0:3072]
    pack_bf16<<<(n + 255) / 256, 256, 0, stream>>>(W_ioux, Wcat16, n);
    n = HID * D_IN;                   // W_fx -> Wcat16[3072:4096]
    pack_bf16<<<(n + 255) / 256, 256, 0, stream>>>(W_fx, Wcat16 + (size_t)THREE_H * D_IN, n);
    n = THREE_H * HID;                // W_iouh (scan)
    pack_bf16<<<(n + 255) / 256, 256, 0, stream>>>(W_iouh, Wh16, n);
    n = HID * HID;                    // W_fh (scan)
    pack_bf16<<<(n + 255) / 256, 256, 0, stream>>>(W_fh, Wf16, n);
    n = N_NODES * D_IN;               // seqs = concat(inputs, edge) -> bf16
    build_seq16<<<(n + 255) / 256, 256, 0, stream>>>(inputs, edge_inputs, seq16, bar);
  }
  // 2) fused input GEMM (WMMA bf16, pre-packed operands)
  {
    dim3 grid(OUT_COLS / 64, N_NODES / 128);
    fused_input_gemm<<<grid, 256, 0, stream>>>(seq16, Wcat16, b_ioux, b_fx, big);
  }
  // 3) persistent sequential scan
  tree_scan<<<SCAN_BLOCKS, SCAN_THREADS, 0, stream>>>(
      big, children, mask, Mch, Wh16, Wf16, b_iouh, b_fh,
      C, H, iou_h, f_pre, bar);
  // 4) root (c, h) -> output
  write_out<<<(2 * HID + 255) / 256, 256, 0, stream>>>(C, H, (float*)d_out);
}
